// PWCFlow_16131897163983
// MI455X (gfx1250) — compile-verified, matrix operand order checked by reading, not run on hardware
//
#include <hip/hip_runtime.h>
#include <math.h>

#define ALPHA   0.1f
#define BATCH   2
#define FEAT_C  32
#define MAX_NCT 18                                 // cinPad <= 576 for this net

typedef __attribute__((ext_vector_type(16))) __bf16        v16bf;
typedef __attribute__((ext_vector_type(8)))  __bf16        v8bf;
typedef __attribute__((ext_vector_type(8)))  float         v8f;
typedef __attribute__((ext_vector_type(4)))  unsigned int  u32x4;

union BfQuad { u32x4 u; v8bf h; };

__device__ __forceinline__ float leakyf(float v) { return v >= 0.f ? v : ALPHA * v; }

__device__ __forceinline__ v16bf cat8(v8bf lo, v8bf hi)
{
    return __builtin_shufflevector(lo, hi, 0,1,2,3,4,5,6,7,8,9,10,11,12,13,14,15);
}

// ---------------------------------------------------------------------------
// Weight pre-pack: w[co][ci][ky][kx] (f32) -> bf16 B-operand panels laid out so
// each lane's 16 values for one (r, ctile, ntile) are 32 contiguous bytes.
// K ordering: k = r*cinPad + ci  (r = ky*3+kx tap index).
// flat = ((r*nCt + ct)*nNt + nt)*512 + lane*16 + e
// ---------------------------------------------------------------------------
__global__ void pack_weights_kernel(const float* __restrict__ w, __bf16* __restrict__ panel,
                                    int cin, int cout, int cinPad, int nNt)
{
    const int nCt = cinPad >> 5;
    const int total = 9 * nCt * nNt * 512;
    int idx = blockIdx.x * blockDim.x + threadIdx.x;
    if (idx >= total) return;
    const int e    = idx & 15;
    const int lane = (idx >> 4) & 31;
    const int nt   = (idx >> 9) % nNt;
    const int ct   = (idx / (512 * nNt)) % nCt;
    const int r    = idx / (512 * nNt * nCt);
    const int half = lane >> 4, l16 = lane & 15;
    const int v = e >> 1, t = e & 1;
    const int kb = (v < 4 ? 2 * v : 16 + 2 * (v - 4)) + half * 8;
    const int ci = ct * 32 + kb + t;
    const int co = nt * 16 + l16;
    float val = 0.f;
    if (ci < cin && co < cout)
        val = w[(size_t)co * cin * 9 + (size_t)ci * 9 + r];   // OIHW, tap index r
    panel[idx] = (__bf16)val;
}

// ---- specialized inner loop: DUAL = two N tiles, MASK = border masking ----
template<bool DUAL, bool MASK>
__device__ __forceinline__ void conv_inner(const __bf16* ap, const __bf16* lb,
                                           int nCt, unsigned int msk,
                                           v8f& acc0, v8f& acc1)
{
    const int bstride = DUAL ? 1024 : 512;
    for (int ct = 0; ct < nCt; ++ct) {
        BfQuad a0, a1;
        a0.u = *(const u32x4*)(ap);
        a1.u = *(const u32x4*)(ap + 16);
        if (MASK) { a0.u = a0.u & msk; a1.u = a1.u & msk; }
        v16bf av = cat8(a0.h, a1.h);
        const __bf16* bp = lb + (size_t)ct * bstride;
        v8bf b00 = *(const v8bf*)(bp);
        v8bf b01 = *(const v8bf*)(bp + 8);
        if (DUAL) {
            v8bf b10 = *(const v8bf*)(bp + 512);
            v8bf b11 = *(const v8bf*)(bp + 520);
            v16bf bv0 = cat8(b00, b01);
            v16bf bv1 = cat8(b10, b11);
            acc0 = __builtin_amdgcn_wmma_f32_16x16x32_bf16(false, av, false, bv0,
                                                           (short)0, acc0, false, false);
            acc1 = __builtin_amdgcn_wmma_f32_16x16x32_bf16(false, av, false, bv1,
                                                           (short)0, acc1, false, false);
        } else {
            v16bf bv0 = cat8(b00, b01);
            acc0 = __builtin_amdgcn_wmma_f32_16x16x32_bf16(false, av, false, bv0,
                                                           (short)0, acc0, false, false);
        }
        ap += 32;
    }
}

// ---------------------------------------------------------------------------
// Implicit-GEMM 3x3 conv (SAME, dilation) on bf16 NHWC activations.
//   M tile  = 16 linearized pixels per wave (8 waves/block)
//   N tiles = 2 x 16 output channels per wave (dual accumulator, shared A)
//   K      = 9 taps x cinPad channels, 32 per WMMA
// Per tap, the block stages its B slice into LDS via global_load_async_to_lds
// (ASYNCcnt + s_wait_asynccnt + split barrier); the inner loop is specialized
// wave-uniformly on {dual-N, border-mask} so the steady state is:
//   2x global_load_b128 (A) + 4x ds_load_b128 (B) + 2x v_wmma.
// M-tile is clamped (duplicate waves redo identical work) so control flow is
// block-uniform at barriers and EXEC is full for every WMMA.
// ---------------------------------------------------------------------------
__global__ __launch_bounds__(256)
void conv3x3_wmma_kernel(const __bf16* __restrict__ src, int Cpad, int cinPad,
                         __bf16* __restrict__ dstB,            // bf16 NHWC (or null)
                         float* __restrict__ dstF,             // f32 NCHW  (or null)
                         int dstCpad, int cout, int H, int W,
                         const __bf16* __restrict__ panel, const float* __restrict__ bias,
                         int dil, int doLeaky, int nNt)
{
    __shared__ __bf16 shB[MAX_NCT * 2 * 512];      // 36 KB: B slices for one tap
    const int lane   = threadIdx.x & 31;
    const int wv     = threadIdx.x >> 5;
    const int hw     = H * W;
    const int numPix = BATCH * hw;                 // multiple of 16
    const int mTiles = numPix >> 4;
    int mTile = blockIdx.x * 8 + wv;
    if (mTile >= mTiles) mTile = mTiles - 1;       // clamp: keeps block uniform
    const int nCt     = cinPad >> 5;
    const int nt0     = blockIdx.y * 2;
    const int nNtLoc  = min(2, nNt - nt0);
    const int half    = lane >> 4;
    const int l16     = lane & 15;
    const int m0      = mTile << 4;

    const int pA  = m0 + l16;
    const int bA  = pA / hw;
    const int rem = pA - bA * hw;
    const int yA  = rem / W;
    const int xA  = rem - yA * W;

    v8f acc0 = {}, acc1 = {};
    const int chunksPerR = nCt * nNtLoc * 64;      // 16-byte chunks per tap

    for (int r = 0; r < 9; ++r) {
        const int ky = r / 3, kx = r - ky * 3;

        // ---- async-stage this tap's B slice (both N tiles) into LDS ----
        __syncthreads();                           // prev tap done with shB
        for (int c = threadIdx.x; c < chunksPerR; c += 256) {
            const int ct  = c / (nNtLoc * 64);
            const int rm  = c - ct * (nNtLoc * 64);
            const int ntl = rm >> 6;
            const int q   = rm & 63;
            const __bf16* gsrc =
                panel + ((size_t)((r * nCt + ct) * nNt + nt0 + ntl)) * 512 + q * 8;
            const unsigned int laddr =
                (unsigned int)(size_t)&shB[((ct * nNtLoc + ntl) * 64 + q) * 8];
            asm volatile("global_load_async_to_lds_b128 %0, %1, off"
                         :: "v"(laddr), "v"((unsigned long long)(size_t)gsrc)
                         : "memory");
        }
        asm volatile("s_wait_asynccnt 0x0" ::: "memory");
        __syncthreads();

        // ---- A addressing for this tap (clamp + mask for SAME padding) ----
        const int iy = yA + (ky - 1) * dil;
        const int ix = xA + (kx - 1) * dil;
        const bool ok = (iy >= 0) & (iy < H) & (ix >= 0) & (ix < W);
        const unsigned int msk = ok ? 0xffffffffu : 0u;
        const bool allOk = (__builtin_amdgcn_ballot_w32(ok) == 0xffffffffu); // uniform
        const int iyc = min(max(iy, 0), H - 1);
        const int ixc = min(max(ix, 0), W - 1);
        const __bf16* ap = src + ((size_t)(bA * hw + iyc * W + ixc)) * Cpad + half * 8;
        const __bf16* lb = shB + (size_t)lane * 16;

        if (nNtLoc == 2) {
            if (allOk) conv_inner<true,  false>(ap, lb, nCt, msk, acc0, acc1);
            else       conv_inner<true,  true >(ap, lb, nCt, msk, acc0, acc1);
        } else {
            if (allOk) conv_inner<false, false>(ap, lb, nCt, msk, acc0, acc1);
            else       conv_inner<false, true >(ap, lb, nCt, msk, acc0, acc1);
        }
    }

    // ---- store both N tiles (duplicate clamped waves write identical data) --
    for (int s = 0; s < nNtLoc; ++s) {
        const int co = (nt0 + s) * 16 + l16;
        if (co >= cout) continue;
        const float bb = bias[co];
        const v8f* accp = s ? &acc1 : &acc0;
#pragma unroll
        for (int i = 0; i < 8; ++i) {
            const int p = m0 + i + 8 * half;       // D: VGPR i -> M=i / M=i+8
            float v = (*accp)[i] + bb;
            if (doLeaky) v = leakyf(v);
            if (dstB) {
                dstB[(size_t)p * dstCpad + co] = (__bf16)v;
            } else {
                const int b = p / hw, rr = p - b * hw;
                const int y = rr / W, x = rr - y * W;
                dstF[((size_t)(b * dstCpad + co)) * hw + y * W + x] = v;
            }
        }
    }
}

// ------------------------- pair normalization ------------------------------
__global__ __launch_bounds__(256)
void meanvar_kernel(const float* __restrict__ f1, const float* __restrict__ f2,
                    int n, float* __restrict__ stats)
{
    const int b = blockIdx.x;
    const float* p1 = f1 + (size_t)b * n;
    const float* p2 = f2 + (size_t)b * n;
    float s1 = 0, q1 = 0, s2 = 0, q2 = 0;
    for (int i = threadIdx.x; i < n; i += blockDim.x) {
        float a = p1[i]; s1 += a; q1 += a * a;
        float c = p2[i]; s2 += c; q2 += c * c;
    }
    __shared__ float sh[4][256];
    sh[0][threadIdx.x] = s1; sh[1][threadIdx.x] = q1;
    sh[2][threadIdx.x] = s2; sh[3][threadIdx.x] = q2;
    __syncthreads();
    for (int off = 128; off > 0; off >>= 1) {
        if (threadIdx.x < off)
            for (int j = 0; j < 4; ++j) sh[j][threadIdx.x] += sh[j][threadIdx.x + off];
        __syncthreads();
    }
    if (threadIdx.x == 0) {
        float inv = 1.f / (float)n;
        float m1 = sh[0][0] * inv, m2 = sh[2][0] * inv;
        float v1 = sh[1][0] * inv - m1 * m1;
        float v2 = sh[3][0] * inv - m2 * m2;
        stats[b * 2 + 0] = 0.5f * (m1 + m2);
        stats[b * 2 + 1] = sqrtf(0.5f * (v1 + v2) + 1e-16f);
    }
}

__global__ void norm_apply_kernel(const float* __restrict__ f1, const float* __restrict__ f2,
                                  const float* __restrict__ stats,
                                  float* __restrict__ f1n, float* __restrict__ f2n,
                                  int n, int total)
{
    int i = blockIdx.x * blockDim.x + threadIdx.x;
    if (i >= total) return;
    int b = i / n;
    float m = stats[b * 2], s = stats[b * 2 + 1];
    f1n[i] = (f1[i] - m) / s;
    f2n[i] = (f2[i] - m) / s;
}

// ---------------- cost volume (f32 NCHW in -> bf16 NHWC out) ---------------
__global__ void cost_volume_kernel(const float* __restrict__ f1n, const float* __restrict__ f2n,
                                   __bf16* __restrict__ bufX, int Cpad, int cvOff, int H, int W)
{
    const int hw = H * W;
    const int total = BATCH * 81 * hw;
    int idx = blockIdx.x * blockDim.x + threadIdx.x;
    if (idx >= total) return;
    int x = idx % W;
    int y = (idx / W) % H;
    int d = (idx / hw) % 81;
    int b = idx / (81 * hw);
    int dy = d / 9 - 4, dx = d % 9 - 4;
    int y2 = y + dy, x2 = x + dx;
    float s = 0.f;
    if (y2 >= 0 && y2 < H && x2 >= 0 && x2 < W) {
        const float* a = f1n + ((size_t)b * FEAT_C) * hw + y * W + x;
        const float* c = f2n + ((size_t)b * FEAT_C) * hw + y2 * W + x2;
        for (int ci = 0; ci < FEAT_C; ++ci) s += a[(size_t)ci * hw] * c[(size_t)ci * hw];
    }
    bufX[((size_t)(b * hw + y * W + x)) * Cpad + cvOff + d] = (__bf16)leakyf(s * (1.f / FEAT_C));
}

// ----------------------------- bilinear warp -------------------------------
__global__ void warp_kernel(const float* __restrict__ f2, const float* __restrict__ flow,
                            float* __restrict__ out, int H, int W)
{
    const int hw = H * W;
    const int total = BATCH * FEAT_C * hw;
    int idx = blockIdx.x * blockDim.x + threadIdx.x;
    if (idx >= total) return;
    int x = idx % W;
    int y = (idx / W) % H;
    int c = (idx / hw) % FEAT_C;
    int b = idx / (FEAT_C * hw);
    float wy_ = (float)y + flow[((size_t)(b * 2 + 0)) * hw + y * W + x];
    float wx_ = (float)x + flow[((size_t)(b * 2 + 1)) * hw + y * W + x];
    float y0f = floorf(wy_), x0f = floorf(wx_);
    int   y0 = (int)y0f, x0 = (int)x0f;
    float fy = wy_ - y0f, fx = wx_ - x0f;
    const float* img = f2 + ((size_t)(b * FEAT_C + c)) * hw;
    float v00 = (y0 >= 0 && y0 < H && x0 >= 0 && x0 < W) ? img[y0 * W + x0] : 0.f;
    float v01 = (y0 >= 0 && y0 < H && x0 + 1 >= 0 && x0 + 1 < W) ? img[y0 * W + x0 + 1] : 0.f;
    float v10 = (y0 + 1 >= 0 && y0 + 1 < H && x0 >= 0 && x0 < W) ? img[(y0 + 1) * W + x0] : 0.f;
    float v11 = (y0 + 1 >= 0 && y0 + 1 < H && x0 + 1 >= 0 && x0 + 1 < W) ? img[(y0 + 1) * W + x0 + 1] : 0.f;
    out[idx] = v00 * (1 - fy) * (1 - fx) + v01 * (1 - fy) * fx
             + v10 * fy * (1 - fx)       + v11 * fy * fx;
}

// ------------------- 2x bilinear flow upsample (x2 scale) ------------------
__global__ void upsample_flow_kernel(const float* __restrict__ in, float* __restrict__ out,
                                     int H, int W)
{
    const int OH = 2 * H, OW = 2 * W;
    const int ohw = OH * OW, hw = H * W;
    const int total = BATCH * 2 * ohw;
    int idx = blockIdx.x * blockDim.x + threadIdx.x;
    if (idx >= total) return;
    int ox = idx % OW;
    int oy = (idx / OW) % OH;
    int c  = (idx / ohw) % 2;
    int b  = idx / (2 * ohw);
    float sy = ((float)oy + 0.5f) * 0.5f - 0.5f;
    float sx = ((float)ox + 0.5f) * 0.5f - 0.5f;
    sy = fminf(fmaxf(sy, 0.f), (float)(H - 1));
    sx = fminf(fmaxf(sx, 0.f), (float)(W - 1));
    int y0 = (int)floorf(sy), x0 = (int)floorf(sx);
    int y1 = min(y0 + 1, H - 1), x1 = min(x0 + 1, W - 1);
    float fy = sy - (float)y0, fx = sx - (float)x0;
    const float* img = in + ((size_t)(b * 2 + c)) * hw;
    float v = img[y0 * W + x0] * (1 - fy) * (1 - fx) + img[y0 * W + x1] * (1 - fy) * fx
            + img[y1 * W + x0] * fy * (1 - fx)       + img[y1 * W + x1] * fy * fx;
    out[idx] = 2.0f * v;
}

// -------- 4x4 stride-2 transposed conv 32->32 (bf16 NHWC in/out) -----------
__global__ void deconv4x4_kernel(const __bf16* __restrict__ src,
                                 const float* __restrict__ wgt, const float* __restrict__ bias,
                                 __bf16* __restrict__ out, int H, int W)
{
    const int OH = 2 * H, OW = 2 * W;
    const int ohw = OH * OW, hw = H * W;
    const int total = BATCH * ohw * FEAT_C;
    int idx = blockIdx.x * blockDim.x + threadIdx.x;
    if (idx >= total) return;
    const int co = idx & 31;
    const int p  = idx >> 5;
    const int ox = p % OW;
    const int oy = (p / OW) % OH;
    const int b  = p / ohw;
    float s = bias[co];
    for (int ky = 0; ky < 4; ++ky) {
        int iy2 = oy + 1 - ky;
        if (iy2 < 0 || (iy2 & 1)) continue;
        int iy = iy2 >> 1;
        if (iy >= H) continue;
        for (int kx = 0; kx < 4; ++kx) {
            int ix2 = ox + 1 - kx;
            if (ix2 < 0 || (ix2 & 1)) continue;
            int ix = ix2 >> 1;
            if (ix >= W) continue;
            const __bf16* sp = src + ((size_t)(b * hw + iy * W + ix)) * FEAT_C;
            const float*  wp = wgt + ((size_t)(ky * 4 + kx) * FEAT_C) * FEAT_C + co;  // HWIO
            for (int ci = 0; ci < FEAT_C; ++ci)
                s += (float)sp[ci] * wp[(size_t)ci * FEAT_C];
        }
    }
    out[(size_t)p * FEAT_C + co] = (__bf16)s;
}

// ------------------------------- utilities ---------------------------------
__global__ void fill_zero_kernel(unsigned int* __restrict__ p, long long n32)
{
    long long i = (long long)blockIdx.x * blockDim.x + threadIdx.x;
    if (i < n32) p[i] = 0u;
}

__global__ void copy_bf16_nhwc_kernel(const __bf16* __restrict__ src, int srcCpad, int srcOff,
                                      __bf16* __restrict__ dst, int dstCpad, int dstOff,
                                      int C, int numPix)
{
    int idx = blockIdx.x * blockDim.x + threadIdx.x;
    if (idx >= numPix * C) return;
    int c = idx % C, p = idx / C;
    dst[(size_t)p * dstCpad + dstOff + c] = src[(size_t)p * srcCpad + srcOff + c];
}

__global__ void cvt_nchw_to_nhwc_kernel(const float* __restrict__ src, int srcCtot, int srcOff,
                                        __bf16* __restrict__ dst, int dstCpad, int dstOff,
                                        int C, int H, int W)
{
    const int hw = H * W;
    const int total = BATCH * C * hw;
    int idx = blockIdx.x * blockDim.x + threadIdx.x;
    if (idx >= total) return;
    int xy = idx % hw;
    int c  = (idx / hw) % C;
    int b  = idx / (C * hw);
    dst[((size_t)(b * hw + xy)) * dstCpad + dstOff + c] =
        (__bf16)src[((size_t)(b * srcCtot + srcOff + c)) * hw + xy];
}

__global__ void add_kernel(float* __restrict__ y, const float* __restrict__ x, int n)
{
    int i = blockIdx.x * blockDim.x + threadIdx.x;
    if (i < n) y[i] += x[i];
}

// ---------------------------------------------------------------------------
static inline int gsz(long long n) { return (int)((n + 255) / 256); }

static inline void conv_launch(const float* w, const float* b,
                               const __bf16* src, int srcCpad, int cin,
                               __bf16* dstB, float* dstF, int dstCpad, int cout,
                               int H, int W, int dil, int leaky,
                               __bf16* panel, hipStream_t s)
{
    const int cinPad = (cin + 31) & ~31;
    const int nCt = cinPad / 32, nNt = (cout + 15) / 16;
    pack_weights_kernel<<<gsz((long long)9 * nCt * nNt * 512), 256, 0, s>>>(
        w, panel, cin, cout, cinPad, nNt);
    const int numPix = BATCH * H * W;
    const int mTiles = numPix / 16;
    dim3 grid((mTiles + 7) / 8, (nNt + 1) / 2);
    conv3x3_wmma_kernel<<<grid, 256, 0, s>>>(src, srcCpad, cinPad, dstB, dstF,
                                             dstCpad, cout, H, W, panel, b, dil, leaky, nNt);
}

extern "C" void kernel_launch(void* const* d_in, const int* in_sizes, int n_in,
                              void* d_out, int out_size, void* d_ws, size_t ws_size,
                              hipStream_t stream)
{
    (void)in_sizes; (void)n_in; (void)out_size; (void)ws_size;

    const float* fp1[4] = {(const float*)d_in[0], (const float*)d_in[1],
                           (const float*)d_in[2], (const float*)d_in[3]};
    const float* fp2[4] = {(const float*)d_in[4], (const float*)d_in[5],
                           (const float*)d_in[6], (const float*)d_in[7]};
    // params flattened in jax pytree order (sorted dict keys): ctx, flow, refine
    const int PCTX = 8, PFLOW = 16, PREF = 64;
    auto ctxW = [&](int lvl) { return (const float*)d_in[PCTX + (lvl - 1) * 2]; };
    auto ctxB = [&](int lvl) { return (const float*)d_in[PCTX + (lvl - 1) * 2 + 1]; };
    auto fW   = [&](int lvl, int j) { return (const float*)d_in[PFLOW + (lvl - 1) * 12 + j * 2]; };
    auto fB   = [&](int lvl, int j) { return (const float*)d_in[PFLOW + (lvl - 1) * 12 + j * 2 + 1]; };
    auto foW  = [&](int lvl) { return (const float*)d_in[PFLOW + (lvl - 1) * 12 + 10]; };
    auto foB  = [&](int lvl) { return (const float*)d_in[PFLOW + (lvl - 1) * 12 + 11]; };
    auto rW   = [&](int j) { return (const float*)d_in[PREF + j * 2]; };
    auto rB   = [&](int j) { return (const float*)d_in[PREF + j * 2 + 1]; };

    // ---- workspace bump allocator (sized for level 1, reused by all levels)
    char* wsBase = (char*)d_ws;
    size_t off = 0;
    auto allocB = [&](size_t bytes) -> void* {
        void* p = (void*)(wsBase + off);
        off += (bytes + 255) & ~(size_t)255;
        return p;
    };
    const size_t hw1 = 256 * 256;
    __bf16* BUFX   = (__bf16*)allocB((size_t)BATCH * 608 * hw1 * 2);   // NHWC concat buffer
    __bf16* CTXBUF = (__bf16*)allocB((size_t)BATCH * 32 * hw1 * 2);    // context (aligned)
    __bf16* CTXUP  = (__bf16*)allocB((size_t)BATCH * 32 * hw1 * 2);    // deconv output
    __bf16* RA     = (__bf16*)allocB((size_t)BATCH * 128 * hw1 * 2);
    __bf16* RB     = (__bf16*)allocB((size_t)BATCH * 128 * hw1 * 2);
    __bf16* PANEL  = (__bf16*)allocB((size_t)512 * 1024 * 2);          // packed weights
    float* WARPED  = (float*)allocB((size_t)BATCH * FEAT_C * hw1 * 4);
    float* F1N     = (float*)allocB((size_t)BATCH * FEAT_C * hw1 * 4);
    float* W2N     = (float*)allocB((size_t)BATCH * FEAT_C * hw1 * 4);
    float* FLOWUP  = (float*)allocB((size_t)BATCH * 2 * hw1 * 4);
    float* REFOUT  = (float*)allocB((size_t)BATCH * 2 * hw1 * 4);
    float* STATS   = (float*)allocB(64);
    float* FLOWL[5];
    for (int lvl = 1; lvl <= 4; ++lvl) {
        int H = 512 >> lvl;
        FLOWL[lvl] = (float*)allocB((size_t)BATCH * 2 * H * H * 4);
    }

    const int DEC_CH[5] = {128, 128, 96, 64, 32};

    for (int lvl = 4; lvl >= 1; --lvl) {
        const int H = 512 >> lvl, W = H;
        const int hw = H * W;
        const int numPix = BATCH * hw;
        const float* f1 = fp1[lvl - 1];
        const float* f2 = fp2[lvl - 1];

        const float* w2;
        if (lvl == 4) {
            w2 = f2;                                   // no flow yet: warped2 == f2
        } else {
            warp_kernel<<<gsz((long long)numPix * FEAT_C), 256, 0, stream>>>(
                f2, FLOWUP, WARPED, H, W);
            w2 = WARPED;
        }

        meanvar_kernel<<<BATCH, 256, 0, stream>>>(f1, w2, FEAT_C * hw, STATS);
        norm_apply_kernel<<<gsz((long long)numPix * FEAT_C), 256, 0, stream>>>(
            f1, w2, STATS, F1N, W2N, FEAT_C * hw, numPix * FEAT_C);

        const int cvOff = (lvl == 4) ? 0 : 34;
        const int cin0  = (lvl == 4) ? 113 : 147;      // [ctx_up, flow_up,] cv, f1
        const int Ctot  = cin0 + 128 + 128 + 96 + 64 + 32;
        const int Cpad  = (Ctot + 31) & ~31;           // 576 / 608

        // zero the level's activation buffer (padding channels must be finite 0)
        fill_zero_kernel<<<gsz((long long)numPix * Cpad / 2), 256, 0, stream>>>(
            (unsigned int*)BUFX, (long long)numPix * Cpad / 2);

        if (lvl < 4) {
            copy_bf16_nhwc_kernel<<<gsz((long long)numPix * FEAT_C), 256, 0, stream>>>(
                CTXUP, FEAT_C, 0, BUFX, Cpad, 0, FEAT_C, numPix);
            cvt_nchw_to_nhwc_kernel<<<gsz((long long)numPix * 2), 256, 0, stream>>>(
                FLOWUP, 2, 0, BUFX, Cpad, 32, 2, H, W);
        }
        cost_volume_kernel<<<gsz((long long)numPix * 81), 256, 0, stream>>>(
            F1N, W2N, BUFX, Cpad, cvOff, H, W);
        cvt_nchw_to_nhwc_kernel<<<gsz((long long)numPix * FEAT_C), 256, 0, stream>>>(
            f1, FEAT_C, 0, BUFX, Cpad, cvOff + 81, FEAT_C, H, W);

        // dense-concat decoder: conv j reads channels [0,cur), appends at cur
        int cur = cin0;
        for (int j = 0; j < 5; ++j) {
            conv_launch(fW(lvl, j), fB(lvl, j), BUFX, Cpad, cur,
                        BUFX + cur, nullptr, Cpad, DEC_CH[j], H, W, 1, 1, PANEL, stream);
            cur += DEC_CH[j];
        }
        // copy context (last 32 channels) to an aligned channel-0 buffer
        copy_bf16_nhwc_kernel<<<gsz((long long)numPix * FEAT_C), 256, 0, stream>>>(
            BUFX, Cpad, cur - 32, CTXBUF, 32, 0, FEAT_C, numPix);
        // flow head: f32 NCHW output, no activation
        conv_launch(foW(lvl), foB(lvl), CTXBUF, 32, 32,
                    nullptr, FLOWL[lvl], 2, 2, H, W, 1, 0, PANEL, stream);

        if (lvl < 4)
            add_kernel<<<gsz((long long)numPix * 2), 256, 0, stream>>>(
                FLOWL[lvl], FLOWUP, numPix * 2);

        if (lvl > 1) {
            upsample_flow_kernel<<<gsz((long long)numPix * 8), 256, 0, stream>>>(
                FLOWL[lvl], FLOWUP, H, W);
            deconv4x4_kernel<<<gsz((long long)numPix * 4 * FEAT_C), 256, 0, stream>>>(
                CTXBUF, ctxW(lvl), ctxB(lvl), CTXUP, H, W);
        }
    }

    // ---------------- refinement (level 1): dilated conv stack ----------------
    {
        const int H = 256, W = 256, hw = H * W;
        const int numPix = BATCH * hw;
        fill_zero_kernel<<<gsz((long long)numPix * 128 / 2), 256, 0, stream>>>(
            (unsigned int*)RA, (long long)numPix * 128 / 2);
        copy_bf16_nhwc_kernel<<<gsz((long long)numPix * FEAT_C), 256, 0, stream>>>(
            CTXBUF, 32, 0, RA, 128, 0, FEAT_C, numPix);
        cvt_nchw_to_nhwc_kernel<<<gsz((long long)numPix * 2), 256, 0, stream>>>(
            FLOWL[1], 2, 0, RA, 128, 32, 2, H, W);

        const int rcin[6]  = {34, 128, 128, 128, 96, 64};
        const int rcout[6] = {128, 128, 128, 96, 64, 32};
        const int rdil[6]  = {1, 2, 4, 8, 16, 1};
        __bf16* pin = RA;
        __bf16* pout = RB;
        for (int j = 0; j < 6; ++j) {
            conv_launch(rW(j), rB(j), pin, 128, rcin[j],
                        pout, nullptr, 128, rcout[j], H, W, rdil[j], 1, PANEL, stream);
            __bf16* t = pin; pin = pout; pout = t;
        }
        conv_launch(rW(6), rB(6), pin, 128, 32,
                    nullptr, REFOUT, 2, 2, H, W, 1, 0, PANEL, stream);
        add_kernel<<<gsz((long long)numPix * 2), 256, 0, stream>>>(
            FLOWL[1], REFOUT, numPix * 2);
    }

    // ---------------- outputs: flows lvl1..lvl4 concatenated ------------------
    float* outF = (float*)d_out;
    size_t o = 0;
    for (int lvl = 1; lvl <= 4; ++lvl) {
        int H = 512 >> lvl;
        size_t n = (size_t)BATCH * 2 * H * H;
        hipMemcpyAsync(outF + o, FLOWL[lvl], n * sizeof(float),
                       hipMemcpyDeviceToDevice, stream);
        o += n;
    }
}
